// GPT2PalsLite_26740466385007
// MI455X (gfx1250) — matrix operand order repaired and checked by reading, not compiled, not run on hardware
//
#include <hip/hip_runtime.h>
#include <stdint.h>

// ---------------- types ----------------
typedef __attribute__((ext_vector_type(16))) __bf16 bf16x16;
typedef __attribute__((ext_vector_type(8)))  float  f32x8;

#define NB    8
#define LSEQ  2048
#define DIN   512
#define DH    200
#define HP    224      // DH padded to multiple of 32 (7 k-chunks)
#define DOUT  512
#define QSCALE 0.07071067811865475f  // 1/sqrt(200)

union FragU { bf16x16 v; uint4 q[2]; unsigned short u[16]; };

static __device__ __forceinline__ unsigned short f2bf(float x) {
  __bf16 h = (__bf16)x;
  return __builtin_bit_cast(unsigned short, h);
}

// A fragment: 16x32 bf16 tile, rows [row..row+15], cols [kk..kk+31] of a
// row-major matrix with leading dim lda (elements).
// ISA 7.12.2: lane<16 -> M=lane, K chunks {8*half..+7, 8*half+16..+23}
static __device__ __forceinline__ bf16x16 load_a(const unsigned short* src, int lda, int row, int kk) {
  const int lane = threadIdx.x & 31;
  const int half = lane >> 4;
  const int l16  = lane & 15;
  const unsigned short* p = src + (size_t)(row + l16) * lda + kk + half * 8;
  FragU f;
  f.q[0] = *(const uint4*)p;
  f.q[1] = *(const uint4*)(p + 16);   // +32 bytes -> K + 16
  return f.v;
}

// B fragment: 32x16 (KxN) tile where memory holds B^T row-major, i.e. src is
// [N, ldb] row-major (weights [n][k], K-matrix [key][h], or Vt [h][key]).
// ISA 7.12.2: lane -> column N=lane&15, K chunks {8*half..+7, 8*half+16..+23}
static __device__ __forceinline__ bf16x16 load_b(const unsigned short* src, int ldb, int nbase, int kk) {
  const int lane = threadIdx.x & 31;
  const int half = lane >> 4;
  const int l16  = lane & 15;
  const unsigned short* p = src + (size_t)(nbase + l16) * ldb + kk + half * 8;
  FragU f;
  f.q[0] = *(const uint4*)p;
  f.q[1] = *(const uint4*)(p + 16);   // +32 bytes -> K + 16
  return f.v;
}

static __device__ __forceinline__ f32x8 wmma_bf16(bf16x16 a, bf16x16 b, f32x8 c) {
  return __builtin_amdgcn_wmma_f32_16x16x32_bf16(false, a, false, b, (short)0, c, false, false);
}

// ---------------- conversion kernels ----------------
__global__ void k_cvt_hs(const float* __restrict__ hs, unsigned short* __restrict__ hsb, int n) {
  int i = blockIdx.x * blockDim.x + threadIdx.x;
  if (i < n) hsb[i] = f2bf(hs[i]);
}

__global__ void k_pack_w(const float* __restrict__ Wq, const float* __restrict__ Wk,
                         const float* __restrict__ Wv, const float* __restrict__ Wo,
                         unsigned short* __restrict__ Wqkvb, unsigned short* __restrict__ Wob) {
  const int nqkv = 3 * HP * DIN;
  int i = blockIdx.x * blockDim.x + threadIdx.x;
  if (i < nqkv) {
    int which = i / (HP * DIN);
    int r = i % (HP * DIN);
    int h = r / DIN, d = r % DIN;
    const float* W = (which == 0) ? Wq : (which == 1) ? Wk : Wv;
    Wqkvb[i] = f2bf(h < DH ? W[h * DIN + d] : 0.f);
  } else {
    int j = i - nqkv;
    if (j < DOUT * HP) {
      int o = j / HP, h = j % HP;
      Wob[j] = f2bf(h < DH ? Wo[o * DH + h] : 0.f);
    }
  }
}

// ---------------- QKV projection: [B*L,512] x [224,512]^T ----------------
// Q, K stored [B*L, HP] bf16 (Q pre-scaled by 1/sqrt(d_h)).
// V stored transposed: Vt[B][HP][LSEQ] bf16, so attention B-fragments for
// ctx += P x V are contiguous 32-byte loads straight from L2.
__global__ void __launch_bounds__(32)
k_qkv(const unsigned short* __restrict__ hsb, const unsigned short* __restrict__ Wqkvb,
      const float* __restrict__ bq, const float* __restrict__ bk, const float* __restrict__ bv,
      unsigned short* __restrict__ Qb, unsigned short* __restrict__ Kb, unsigned short* __restrict__ Vt) {
  const int rowbase = blockIdx.x * 16;     // over B*L
  const int ntile   = blockIdx.y;          // 0..13
  const int which   = blockIdx.z;          // 0=q 1=k 2=v
  const unsigned short* W = Wqkvb + (size_t)which * HP * DIN;

  f32x8 acc = {0.f,0.f,0.f,0.f,0.f,0.f,0.f,0.f};
  #pragma unroll 4
  for (int kk = 0; kk < DIN; kk += 32) {
    bf16x16 a = load_a(hsb, DIN, rowbase, kk);
    bf16x16 b = load_b(W,   DIN, ntile * 16, kk);
    acc = wmma_bf16(a, b, acc);
  }
  const int lane = threadIdx.x & 31, half = lane >> 4, l16 = lane & 15;
  const int n = ntile * 16 + l16;
  const float* bias = (which == 0) ? bq : (which == 1) ? bk : bv;
  const float bn  = (n < DH) ? bias[n] : 0.f;

  if (which == 2) {
    // transposed store: Vt[b][n][l..]
    const int b  = rowbase >> 11;         // / LSEQ
    const int l  = rowbase & (LSEQ - 1);
    unsigned short* vrow = Vt + ((size_t)b * HP + n) * LSEQ + l;
    #pragma unroll
    for (int r = 0; r < 8; ++r) {
      int m = r + half * 8;
      vrow[m] = f2bf(acc[r] + bn);
    }
  } else {
    const float scl = (which == 0) ? QSCALE : 1.f;   // fold 1/sqrt(d_h) into Q
    unsigned short* out = (which == 0) ? Qb : Kb;
    #pragma unroll
    for (int r = 0; r < 8; ++r) {
      int m = r + half * 8;
      out[(size_t)(rowbase + m) * HP + n] = f2bf((acc[r] + bn) * scl);
    }
  }
}

// ---------------- flash attention: one wave per 16-query tile ----------------
__global__ void __launch_bounds__(32)
k_attn(const unsigned short* __restrict__ Qb, const unsigned short* __restrict__ Kb,
       const unsigned short* __restrict__ Vt, const int* __restrict__ mask,
       unsigned short* __restrict__ Ctxb) {
  __shared__ unsigned short Plds[16 * 32];   // P tile, row-major [16 q][32 key]

  const int qt = blockIdx.x;                 // 0..127
  const int b  = blockIdx.y;                 // 0..7
  const int lane = threadIdx.x & 31, half = lane >> 4, l16 = lane & 15;
  const int qrow = b * LSEQ + qt * 16;       // row into flattened [B*L, HP]
  const unsigned short* Kbase  = Kb + (size_t)b * LSEQ * HP;
  const unsigned short* VtBase = Vt + (size_t)b * HP * LSEQ;   // [HP][LSEQ]
  const int* mrow = mask + ((size_t)b * LSEQ + qt * 16) * LSEQ;

  f32x8 acc[14];
  #pragma unroll
  for (int t = 0; t < 14; ++t) acc[t] = (f32x8){0.f,0.f,0.f,0.f,0.f,0.f,0.f,0.f};
  float mi[8], li[8];
  #pragma unroll
  for (int r = 0; r < 8; ++r) { mi[r] = -3.0e38f; li[r] = 0.f; }

  for (int kb = 0; kb < LSEQ; kb += 32) {
    // ---- scores S[16 q][32 key] as two 16x16 f32 fragments ----
    f32x8 s0 = {0.f,0.f,0.f,0.f,0.f,0.f,0.f,0.f};
    f32x8 s1 = {0.f,0.f,0.f,0.f,0.f,0.f,0.f,0.f};
    #pragma unroll
    for (int kc = 0; kc < HP; kc += 32) {
      bf16x16 a  = load_a(Qb,    HP, qrow,    kc);
      bf16x16 b0 = load_b(Kbase, HP, kb,      kc);
      bf16x16 b1 = load_b(Kbase, HP, kb + 16, kc);
      s0 = wmma_bf16(a, b0, s0);
      s1 = wmma_bf16(a, b1, s1);
    }

    // ---- mask + row max (C layout: lane = key col, vgpr r = row r+8*half) ----
    const int key0 = kb + l16, key1 = kb + 16 + l16;
    float mx[8];
    #pragma unroll
    for (int r = 0; r < 8; ++r) {
      int m = r + half * 8;
      const int* mp = mrow + (size_t)m * LSEQ;
      float v0 = s0[r], v1 = s1[r];
      if (mp[key0] == 0) v0 = -1e20f;
      if (mp[key1] == 0) v1 = -1e20f;
      s0[r] = v0; s1[r] = v1;
      mx[r] = fmaxf(v0, v1);
      // prefetch next key block's mask line for this row (128B = 32 ints)
      if (kb + 32 < LSEQ) __builtin_prefetch(mp + key0 + 32, 0, 1);
    }
    #pragma unroll
    for (int off = 1; off <= 8; off <<= 1) {
      #pragma unroll
      for (int r = 0; r < 8; ++r)
        mx[r] = fmaxf(mx[r], __shfl_xor(mx[r], off, 32));
    }

    // ---- online softmax update ----
    float fac[8], ps[8];
    #pragma unroll
    for (int r = 0; r < 8; ++r) {
      float mn = fmaxf(mi[r], mx[r]);
      fac[r] = __expf(mi[r] - mn);
      mi[r]  = mn;
      s0[r] = __expf(s0[r] - mn);   // reuse s0/s1 as P
      s1[r] = __expf(s1[r] - mn);
      ps[r] = s0[r] + s1[r];
    }
    #pragma unroll
    for (int off = 1; off <= 8; off <<= 1) {
      #pragma unroll
      for (int r = 0; r < 8; ++r)
        ps[r] += __shfl_xor(ps[r], off, 32);
    }
    #pragma unroll
    for (int r = 0; r < 8; ++r) li[r] = li[r] * fac[r] + ps[r];
    #pragma unroll
    for (int t = 0; t < 14; ++t)
      #pragma unroll
      for (int r = 0; r < 8; ++r) acc[t][r] *= fac[r];

    // ---- P -> LDS (bf16, row-major [16][32]) for C-layout -> A-layout ----
    #pragma unroll
    for (int r = 0; r < 8; ++r) {
      int m = r + half * 8;
      Plds[m * 32 + l16]      = f2bf(s0[r]);
      Plds[m * 32 + 16 + l16] = f2bf(s1[r]);
    }
    __syncthreads();

    // ---- ctx += P[16,32] x V[32, HP]; V B-frags contiguous from Vt ----
    bf16x16 pa = load_a(Plds, 32, 0, 0);
    #pragma unroll
    for (int t = 0; t < 14; ++t) {
      bf16x16 vb = load_b(VtBase, LSEQ, t * 16, kb);
      acc[t] = wmma_bf16(pa, vb, acc[t]);
    }
    __syncthreads();
  }

  // ---- epilogue: normalize by row sums, store bf16 context ----
  #pragma unroll
  for (int r = 0; r < 8; ++r) li[r] = 1.f / li[r];
  #pragma unroll
  for (int t = 0; t < 14; ++t) {
    #pragma unroll
    for (int r = 0; r < 8; ++r) {
      int m = r + half * 8;
      Ctxb[(size_t)(qrow + m) * HP + t * 16 + l16] = f2bf(acc[t][r] * li[r]);
    }
  }
}

// ---------------- output projection: [B*L,224] x [512,224]^T + bo -> f32 ----------------
__global__ void __launch_bounds__(32)
k_oproj(const unsigned short* __restrict__ Ctxb, const unsigned short* __restrict__ Wob,
        const float* __restrict__ bo, float* __restrict__ out) {
  const int rowbase = blockIdx.x * 16;
  const int ntile   = blockIdx.y;          // 0..31
  f32x8 acc = {0.f,0.f,0.f,0.f,0.f,0.f,0.f,0.f};
  #pragma unroll
  for (int kk = 0; kk < HP; kk += 32) {
    bf16x16 a = load_a(Ctxb, HP, rowbase, kk);
    bf16x16 b = load_b(Wob,  HP, ntile * 16, kk);
    acc = wmma_bf16(a, b, acc);
  }
  const int lane = threadIdx.x & 31, half = lane >> 4, l16 = lane & 15;
  const int n = ntile * 16 + l16;
  const float bn = bo[n];
  #pragma unroll
  for (int r = 0; r < 8; ++r) {
    int m = r + half * 8;
    out[(size_t)(rowbase + m) * DOUT + n] = acc[r] + bn;
  }
}

// ---------------- launch ----------------
extern "C" void kernel_launch(void* const* d_in, const int* in_sizes, int n_in,
                              void* d_out, int out_size, void* d_ws, size_t ws_size,
                              hipStream_t stream) {
  const float* hs   = (const float*)d_in[0];
  const int*   mask = (const int*)  d_in[1];
  const float* Wq   = (const float*)d_in[2];
  const float* bq   = (const float*)d_in[3];
  const float* Wk   = (const float*)d_in[4];
  const float* bk   = (const float*)d_in[5];
  const float* Wv   = (const float*)d_in[6];
  const float* bv   = (const float*)d_in[7];
  const float* Wo   = (const float*)d_in[8];
  const float* bo   = (const float*)d_in[9];
  float* out = (float*)d_out;

  char* ws = (char*)d_ws;
  size_t off = 0;
  auto wsalloc = [&](size_t bytes) -> void* {
    void* p = ws + off;
    off += (bytes + 255) & ~(size_t)255;
    return p;
  };
  unsigned short* hsb   = (unsigned short*)wsalloc((size_t)NB * LSEQ * DIN * 2);
  unsigned short* Wqkvb = (unsigned short*)wsalloc((size_t)3 * HP * DIN * 2);
  unsigned short* Wob   = (unsigned short*)wsalloc((size_t)DOUT * HP * 2);
  unsigned short* Qb    = (unsigned short*)wsalloc((size_t)NB * LSEQ * HP * 2);
  unsigned short* Kb    = (unsigned short*)wsalloc((size_t)NB * LSEQ * HP * 2);
  unsigned short* Vt    = (unsigned short*)wsalloc((size_t)NB * HP * LSEQ * 2);
  unsigned short* Ctxb  = (unsigned short*)wsalloc((size_t)NB * LSEQ * HP * 2);
  (void)ws_size; (void)in_sizes; (void)n_in; (void)out_size;

  // 1) convert hidden states to bf16
  {
    int n = NB * LSEQ * DIN;
    k_cvt_hs<<<(n + 255) / 256, 256, 0, stream>>>(hs, hsb, n);
  }
  // 2) pack padded bf16 weights
  {
    int n = 3 * HP * DIN + DOUT * HP;
    k_pack_w<<<(n + 255) / 256, 256, 0, stream>>>(Wq, Wk, Wv, Wo, Wqkvb, Wob);
  }
  // 3) QKV projections (V stored transposed)
  {
    dim3 g((NB * LSEQ) / 16, HP / 16, 3);
    k_qkv<<<g, 32, 0, stream>>>(hsb, Wqkvb, bq, bk, bv, Qb, Kb, Vt);
  }
  // 4) flash attention
  {
    dim3 g(LSEQ / 16, NB);
    k_attn<<<g, 32, 0, stream>>>(Qb, Kb, Vt, mask, Ctxb);
  }
  // 5) output projection
  {
    dim3 g((NB * LSEQ) / 16, DOUT / 16);
    k_oproj<<<g, 32, 0, stream>>>(Ctxb, Wob, bo, out);
  }
}
